// VicinityVisionAttention_40596030882114
// MI455X (gfx1250) — matrix operand order repaired and checked by reading, not compiled, not run on hardware
//
#include <hip/hip_runtime.h>

typedef __attribute__((ext_vector_type(16))) __bf16 v16bf;
typedef __attribute__((ext_vector_type(8)))  __bf16 v8bf;
typedef __attribute__((ext_vector_type(8)))  float  v8f;

#define BB 16
#define NN 4096
#define CC 512
#define CF 256
#define NH 8
#define HD 32

__device__ inline v8f zero8() {
  v8f z;
  #pragma unroll
  for (int i = 0; i < 8; ++i) z[i] = 0.0f;
  return z;
}

__device__ inline v8f wmma_bf16(v16bf a, v16bf b, v8f c) {
  return __builtin_amdgcn_wmma_f32_16x16x32_bf16(false, a, false, b, (short)0, c, false, false);
}

__device__ inline float abs_clamp(float t) {
  if (t == 0.0f) return 0.0f;
  float a = fminf(fmaxf(fabsf(t), 1e-4f), 1e4f);
  return copysignf(a, t);
}

// cosFormer position weight w_j(n)
__device__ inline float wjval(int n, int j, int H_, int W_) {
  const float c = 1.5707963267948966f;
  int r = n / W_, q = n - r * W_;
  float ca = c * (float)r / (float)H_;
  float cb = c * (float)q / (float)W_;
  if (j == 0) return __cosf(ca);
  if (j == 1) return __sinf(ca);
  if (j == 2) return __cosf(cb);
  return __sinf(cb);
}

// A fragment: 16x32 bf16, row-major, K contiguous, leading dim ld
__device__ inline v16bf load_a(const __bf16* base, int ld) {
  int lane = threadIdx.x & 31;
  int m = lane & 15;
  int kb = (lane >> 4) << 3;  // 0 or 8
  const __bf16* p = base + m * ld + kb;
  union { v16bf v; v8bf h[2]; } u;
  u.h[0] = *(const v8bf*)(p);
  u.h[1] = *(const v8bf*)(p + 16);
  return u.v;
}

// B fragment: 32x16 bf16, column n has 32 contiguous K values, leading dim ld
__device__ inline v16bf load_b(const __bf16* base, int ld) {
  int lane = threadIdx.x & 31;
  int n = lane & 15;
  int kb = (lane >> 4) << 3;  // 0 or 8
  const __bf16* p = base + n * ld + kb;
  union { v16bf v; v8bf h[2]; } u;
  u.h[0] = *(const v8bf*)(p);
  u.h[1] = *(const v8bf*)(p + 16);
  return u.v;
}

// A fragment scaled per K element (scale indexed by local K)
__device__ inline v16bf load_a_kscale(const __bf16* base, int ld, const float* wk) {
  int lane = threadIdx.x & 31;
  int m = lane & 15;
  int kb = (lane >> 4) << 3;
  const __bf16* p = base + m * ld;
  v16bf f;
  #pragma unroll
  for (int e = 0; e < 8; ++e) {
    f[e]     = (__bf16)((float)p[kb + e]      * wk[kb + e]);
    f[e + 8] = (__bf16)((float)p[kb + 16 + e] * wk[kb + 16 + e]);
  }
  return f;
}

// A fragment scaled per row m (scale = wm4[m*4+j])
__device__ inline v16bf load_a_rowscale(const __bf16* base, int ld, const float* wm4, int j) {
  int lane = threadIdx.x & 31;
  int m = lane & 15;
  int kb = (lane >> 4) << 3;
  float s = wm4[m * 4 + j];
  const __bf16* p = base + m * ld + kb;
  v16bf f;
  #pragma unroll
  for (int e = 0; e < 8; ++e) {
    f[e]     = (__bf16)((float)p[e] * s);
    f[e + 8] = (__bf16)((float)p[16 + e] * s);
  }
  return f;
}

// ---------------- kernel 0: weight fp32 -> bf16 -----------------
__global__ void k_convert(const float* Wq, const float* Wk, const float* Wv, const float* Wo,
                          __bf16* wq, __bf16* wk, __bf16* wv, __bf16* wo) {
  int idx = blockIdx.x * 256 + threadIdx.x;  // 131072 elements per matrix
  const float* src; __bf16* dst;
  switch (blockIdx.y) {
    case 0: src = Wq; dst = wq; break;
    case 1: src = Wk; dst = wk; break;
    case 2: src = Wv; dst = wv; break;
    default: src = Wo; dst = wo; break;
  }
  dst[idx] = (__bf16)src[idx];
}

// ---------------- kernel 1: SE mean (sum) -----------------
__global__ void k_se_reduce(const float* __restrict__ query, float* __restrict__ se_mean) {
  int b = blockIdx.z;
  int c = blockIdx.x * 256 + threadIdx.x;
  int n0 = blockIdx.y * 512;
  const float* p = query + ((size_t)b * NN + n0) * CC + c;
  float s = 0.0f;
  for (int i = 0; i < 512; ++i) s += p[(size_t)i * CC];
  atomicAdd(&se_mean[b * CC + c], s);
}

// ---------------- kernel 2: SE MLP -> sigmoid gate -----------------
__global__ __launch_bounds__(256) void k_se_mlp(const float* se_mean, const float* Wse1,
                                                const float* Wse2, float* se_gate) {
  __shared__ float h[256];
  int b = blockIdx.x, t = threadIdx.x;
  const float* m = se_mean + b * CC;
  const float* w1 = Wse1 + (size_t)t * CC;
  float s = 0.0f;
  for (int c = 0; c < CC; ++c) s += (m[c] * (1.0f / (float)NN)) * w1[c];
  h[t] = fmaxf(s, 0.0f);
  __syncthreads();
  #pragma unroll
  for (int o = 0; o < 2; ++o) {
    int c = t + o * 256;
    const float* w2 = Wse2 + (size_t)c * 256;
    float g = 0.0f;
    for (int j = 0; j < 256; ++j) g += h[j] * w2[j];
    se_gate[b * CC + c] = 1.0f / (1.0f + __expf(-g));
  }
}

// ---------------- kernel 3: fused q/k/v projection (WMMA) -----------------
__global__ __launch_bounds__(128) void k_proj(const float* __restrict__ query,
    const __bf16* wq, const __bf16* wk, const __bf16* wv,
    const float* bq, const float* bk, const float* bv,
    __bf16* qo, __bf16* ko, __bf16* vo) {
  __shared__ __bf16 qtile[16 * 512];
  __shared__ __bf16 otile[16 * 256];
  int tid = threadIdx.x;
  size_t r0 = (size_t)blockIdx.x * 16;  // global token tile over B*N
  // stage 16x512 fp32 -> bf16 LDS
  #pragma unroll
  for (int i = 0; i < 16; ++i) {
    int idx = tid + i * 128;  // float4 index, 2048 total
    int row = idx >> 7, cv = idx & 127;
    float4 f4 = *(const float4*)(query + (r0 + row) * CC + cv * 4);
    __bf16* d = qtile + row * 512 + cv * 4;
    d[0] = (__bf16)f4.x; d[1] = (__bf16)f4.y; d[2] = (__bf16)f4.z; d[3] = (__bf16)f4.w;
  }
  __syncthreads();
  int mat = blockIdx.y;
  const __bf16* W = (mat == 0) ? wq : (mat == 1) ? wk : wv;
  const float* bias = (mat == 0) ? bq : (mat == 1) ? bk : bv;
  int wave = tid >> 5, lane = tid & 31;
  int col0 = wave * 64;
  v8f acc[4];
  #pragma unroll
  for (int t = 0; t < 4; ++t) acc[t] = zero8();
  for (int k0 = 0; k0 < 512; k0 += 32) {
    v16bf a = load_a(qtile + k0, 512);
    #pragma unroll
    for (int t = 0; t < 4; ++t) {
      v16bf bfr = load_b(W + (size_t)(col0 + t * 16) * 512 + k0, 512);
      acc[t] = wmma_bf16(a, bfr, acc[t]);
    }
  }
  int n_ = lane & 15, mb = (lane >> 4) * 8;
  #pragma unroll
  for (int t = 0; t < 4; ++t) {
    int col = col0 + t * 16 + n_;
    float bsv = bias[col];
    #pragma unroll
    for (int r = 0; r < 8; ++r) {
      float v = acc[t][r] + bsv;
      if (mat < 2) v = fmaxf(v, 0.0f);  // relu for q,k only
      otile[(mb + r) * 256 + col] = (__bf16)v;
    }
  }
  __syncthreads();
  __bf16* dst = (mat == 0) ? qo : (mat == 1) ? ko : vo;
  #pragma unroll
  for (int i = 0; i < 4; ++i) {
    int idx = tid * 32 + i * 8;
    int row = idx >> 8, c = idx & 255;
    *(v8bf*)(dst + (r0 + row) * CF + c) = *(const v8bf*)(otile + row * 256 + c);
  }
}

// ---------------- kernel 4: kv state  kv_j = sum_n k~_j(n)^T v(n) -----------------
__global__ __launch_bounds__(128) void k_kv(const __bf16* __restrict__ kbf,
                                            const __bf16* __restrict__ vbf,
                                            __bf16* kvT, const int* Hp, const int* Wp) {
  __shared__ __bf16 kT[32 * 32];  // [f][n]
  __shared__ __bf16 vT[32 * 32];  // [d][n]
  __shared__ float wj[4 * 32];    // [j][n]
  int tid = threadIdx.x, wave = tid >> 5, lane = tid & 31;
  int bh = blockIdx.x, b = bh >> 3, h = bh & 7;
  int H_ = Hp[0], W_ = Wp[0];
  v8f acc[4];
  #pragma unroll
  for (int t = 0; t < 4; ++t) acc[t] = zero8();
  int i = tid >> 2, f0 = (tid & 3) * 8;
  for (int n0 = 0; n0 < NN; n0 += 32) {
    size_t rb = ((size_t)(b * NN + n0 + i)) * CF + h * HD + f0;
    v8bf kk = *(const v8bf*)(kbf + rb);
    v8bf vv = *(const v8bf*)(vbf + rb);
    #pragma unroll
    for (int ff = 0; ff < 8; ++ff) { kT[(f0 + ff) * 32 + i] = kk[ff]; vT[(f0 + ff) * 32 + i] = vv[ff]; }
    { int jj = tid >> 5, nn = tid & 31; wj[jj * 32 + nn] = wjval(n0 + nn, jj, H_, W_); }
    __syncthreads();
    const float* wr = wj + wave * 32;
    v16bf a0 = load_a_kscale(kT, 32, wr);
    v16bf a1 = load_a_kscale(kT + 16 * 32, 32, wr);
    v16bf b0 = load_b(vT, 32);
    v16bf b1 = load_b(vT + 16 * 32, 32);
    acc[0] = wmma_bf16(a0, b0, acc[0]);
    acc[1] = wmma_bf16(a0, b1, acc[1]);
    acc[2] = wmma_bf16(a1, b0, acc[2]);
    acc[3] = wmma_bf16(a1, b1, acc[3]);
    __syncthreads();
  }
  // store transposed [d][f], clamped, bf16:  kvT[b][h][j=wave][d][f]
  __bf16* base = kvT + ((size_t)(bh * 4 + wave)) * 1024;
  int n_ = lane & 15, mb = (lane >> 4) * 8;
  #pragma unroll
  for (int mt = 0; mt < 2; ++mt) {
    #pragma unroll
    for (int nt = 0; nt < 2; ++nt) {
      int d = nt * 16 + n_;
      #pragma unroll
      for (int r = 0; r < 8; ++r) {
        int f = mt * 16 + mb + r;
        base[d * 32 + f] = (__bf16)abs_clamp(acc[mt * 2 + nt][r]);
      }
    }
  }
}

// ---------------- kernel 5: ksum_j[f] = sum_n w_j(n) k[n,f] -----------------
__global__ __launch_bounds__(128) void k_ksum(const __bf16* __restrict__ kbf, float* ksum,
                                              const int* Hp, const int* Wp) {
  int bh = blockIdx.x, b = bh >> 3, h = bh & 7;
  int jj = threadIdx.x >> 5, f = threadIdx.x & 31;
  int H_ = Hp[0], W_ = Wp[0];
  int n0 = blockIdx.y * 512;
  float s = 0.0f;
  for (int nn = 0; nn < 512; ++nn) {
    int n = n0 + nn;
    s += wjval(n, jj, H_, W_) * (float)kbf[((size_t)(b * NN + n)) * CF + h * HD + f];
  }
  atomicAdd(&ksum[(bh * 4 + jj) * 32 + f], s);
}

// ---------------- kernel 6: fused attention lookup + z + out-proj + SE gate ----
__global__ __launch_bounds__(128) void k_out(const __bf16* __restrict__ qbf,
    const __bf16* __restrict__ kvT, const float* __restrict__ ksum,
    const __bf16* __restrict__ wo, const float* __restrict__ bo,
    const float* __restrict__ se_gate, float* __restrict__ out,
    const int* Hp, const int* Wp) {
  __shared__ __bf16 qs[16 * 256];
  __shared__ __bf16 att[16 * 256];
  __shared__ float wrow[16 * 4];
  __shared__ float zs[16 * 8];
  __shared__ float outs[16 * 512];
  int tid = threadIdx.x, wave = tid >> 5, lane = tid & 31;
  size_t r0 = (size_t)blockIdx.x * 16;
  int b = (int)(r0 >> 12);
  int nb = (int)(r0 & (NN - 1));
  int H_ = Hp[0], W_ = Wp[0];
  #pragma unroll
  for (int i = 0; i < 4; ++i) {
    int idx = tid * 32 + i * 8;
    int row = idx >> 8, c = idx & 255;
    *(v8bf*)(qs + row * 256 + c) = *(const v8bf*)(qbf + (r0 + row) * CF + c);
  }
  if (tid < 64) { int m = tid >> 2, jj = tid & 3; wrow[m * 4 + jj] = wjval(nb + m, jj, H_, W_); }
  __syncthreads();
  // z = clamp(1/(q_ . ksum + eps))
  { int m = tid >> 3, hh = tid & 7;
    const float* ks = ksum + (size_t)((b * NH + hh) * 4) * 32;
    float s = 0.0f;
    #pragma unroll
    for (int jj = 0; jj < 4; ++jj) {
      float sj = 0.0f;
      for (int f = 0; f < 32; ++f) sj += (float)qs[m * 256 + hh * HD + f] * ks[jj * 32 + f];
      s += wrow[m * 4 + jj] * sj;
    }
    zs[m * 8 + hh] = abs_clamp(1.0f / (s + 1e-4f));
  }
  __syncthreads();
  // att = clamp(clamp(q_ @ kv) * z)
  int n_ = lane & 15, mb = (lane >> 4) * 8;
  #pragma unroll
  for (int hi = 0; hi < 2; ++hi) {
    int h = wave * 2 + hi;
    v8f a0 = zero8(), a1 = zero8();
    #pragma unroll
    for (int jj = 0; jj < 4; ++jj) {
      v16bf a = load_a_rowscale(qs + h * HD, 256, wrow, jj);
      const __bf16* kb2 = kvT + ((size_t)((b * NH + h) * 4 + jj)) * 1024;
      v16bf b0 = load_b(kb2, 32);
      v16bf b1 = load_b(kb2 + 16 * 32, 32);
      a0 = wmma_bf16(a, b0, a0);
      a1 = wmma_bf16(a, b1, a1);
    }
    #pragma unroll
    for (int nt = 0; nt < 2; ++nt) {
      int d = nt * 16 + n_;
      #pragma unroll
      for (int r = 0; r < 8; ++r) {
        int m = mb + r;
        float v = abs_clamp(nt ? a1[r] : a0[r]);
        v = abs_clamp(v * zs[m * 8 + h]);
        att[m * 256 + h * HD + d] = (__bf16)v;
      }
    }
  }
  __syncthreads();
  // final projection: att @ Wo^T + bo, then SE residual gating
  int c0 = wave * 128;
  v8f acc3[8];
  #pragma unroll
  for (int t = 0; t < 8; ++t) acc3[t] = zero8();
  for (int k0 = 0; k0 < 256; k0 += 32) {
    v16bf a = load_a(att + k0, 256);
    #pragma unroll
    for (int t = 0; t < 8; ++t) {
      v16bf bfr = load_b(wo + (size_t)(c0 + t * 16) * 256 + k0, 256);
      acc3[t] = wmma_bf16(a, bfr, acc3[t]);
    }
  }
  #pragma unroll
  for (int t = 0; t < 8; ++t) {
    int col = c0 + t * 16 + n_;
    float bsv = bo[col];
    float g = se_gate[b * CC + col];
    #pragma unroll
    for (int r = 0; r < 8; ++r) {
      float v = abs_clamp(acc3[t][r] + bsv);
      v = abs_clamp(v + v * g);
      outs[(mb + r) * 512 + col] = v;
    }
  }
  __syncthreads();
  #pragma unroll
  for (int i = 0; i < 16; ++i) {
    int idx = tid * 64 + i * 4;
    int row = idx >> 9, c = idx & 511;
    *(float4*)(out + (r0 + row) * CC + c) = *(const float4*)(outs + row * 512 + c);
  }
}

extern "C" void kernel_launch(void* const* d_in, const int* in_sizes, int n_in,
                              void* d_out, int out_size, void* d_ws, size_t ws_size,
                              hipStream_t stream) {
  (void)in_sizes; (void)n_in; (void)out_size; (void)ws_size;
  const float* query = (const float*)d_in[0];
  const float* Wq   = (const float*)d_in[1];
  const float* bq   = (const float*)d_in[2];
  const float* Wk   = (const float*)d_in[3];
  const float* bk   = (const float*)d_in[4];
  const float* Wv   = (const float*)d_in[5];
  const float* bv   = (const float*)d_in[6];
  const float* Wo   = (const float*)d_in[7];
  const float* bo   = (const float*)d_in[8];
  const float* Wse1 = (const float*)d_in[9];
  const float* Wse2 = (const float*)d_in[10];
  const int*   Hp   = (const int*)d_in[11];
  const int*   Wp   = (const int*)d_in[12];
  float* out = (float*)d_out;

  char* ws = (char*)d_ws;
  size_t off = 0;
  auto alloc = [&](size_t bytes) -> void* {
    void* p = ws + off;
    off = (off + bytes + 255) & ~(size_t)255;
    return p;
  };
  __bf16* wq_bf = (__bf16*)alloc((size_t)CF * CC * 2);
  __bf16* wk_bf = (__bf16*)alloc((size_t)CF * CC * 2);
  __bf16* wv_bf = (__bf16*)alloc((size_t)CF * CC * 2);
  __bf16* wo_bf = (__bf16*)alloc((size_t)CC * CF * 2);
  __bf16* q_bf  = (__bf16*)alloc((size_t)BB * NN * CF * 2);
  __bf16* k_bf  = (__bf16*)alloc((size_t)BB * NN * CF * 2);
  __bf16* v_bf  = (__bf16*)alloc((size_t)BB * NN * CF * 2);
  __bf16* kvT   = (__bf16*)alloc((size_t)BB * NH * 4 * 32 * 32 * 2);
  float* ksum    = (float*)alloc((size_t)BB * NH * 4 * 32 * 4);
  float* se_mean = (float*)alloc((size_t)BB * CC * 4);
  float* se_gate = (float*)alloc((size_t)BB * CC * 4);

  (void)hipMemsetAsync(se_mean, 0, (size_t)BB * CC * 4, stream);
  (void)hipMemsetAsync(ksum, 0, (size_t)BB * NH * 4 * 32 * 4, stream);

  k_convert<<<dim3(512, 4), 256, 0, stream>>>(Wq, Wk, Wv, Wo, wq_bf, wk_bf, wv_bf, wo_bf);
  k_se_reduce<<<dim3(2, 8, BB), 256, 0, stream>>>(query, se_mean);
  k_se_mlp<<<BB, 256, 0, stream>>>(se_mean, Wse1, Wse2, se_gate);
  k_proj<<<dim3((BB * NN) / 16, 3), 128, 0, stream>>>(query, wq_bf, wk_bf, wv_bf,
                                                      bq, bk, bv, q_bf, k_bf, v_bf);
  k_kv<<<BB * NH, 128, 0, stream>>>(k_bf, v_bf, kvT, Hp, Wp);
  k_ksum<<<dim3(BB * NH, 8), 128, 0, stream>>>(k_bf, ksum, Hp, Wp);
  k_out<<<(BB * NN) / 16, 128, 0, stream>>>(q_bf, kvT, ksum, wo_bf, bo, se_gate, out, Hp, Wp);
}
